// MyModel_70171175682302
// MI455X (gfx1250) — matrix operand order repaired and checked by reading, not compiled, not run on hardware
//
#include <hip/hip_runtime.h>
#include <hip/hip_bf16.h>
#include <string.h>

// ---------------------------------------------------------------------------
// MI455X (gfx1250) implementation.
// Everything fits in the 192MB L2 -> conv stack and the two K=21504 GEMMs are
// matrix-throughput bound -> bf16 WMMA (f32 accumulate) everywhere.
// Weights are pre-cast once to bf16 in a K-padded layout so the WMMA inner
// loop is pure vector loads + v_wmma. Implicit-GEMM conv is templated on
// (KH,KW,PAD,MT): compile-time im2col divisors, and MT 16-row M-tiles per
// wave (MT=8 on all heavy layers) so one im2col B-fragment feeds MT v_wmma.
// ---------------------------------------------------------------------------

typedef __bf16 bf16_t;
typedef __attribute__((ext_vector_type(16))) __bf16 v16bf;
typedef __attribute__((ext_vector_type(8)))  __bf16 v8bf;
typedef __attribute__((ext_vector_type(8)))  float  v8f;

#define BSZ   128
#define TSTEP 6
#define BT    768          // B * T
#define NPCAP 1344         // 6*7*32 primary capsules
#define HID   256

__device__ __forceinline__ bf16_t f2bf(float f) {
  union { float f; unsigned u; } v; v.f = f;
  unsigned r = (v.u + 0x7FFFu + ((v.u >> 16) & 1u)) >> 16;   // RNE
  unsigned short h = (unsigned short)r;
  bf16_t o; __builtin_memcpy(&o, &h, 2); return o;
}
__device__ __forceinline__ float bf2f(bf16_t b) {
  unsigned short h; __builtin_memcpy(&h, &b, 2);
  union { float f; unsigned u; } v; v.u = ((unsigned)h) << 16; return v.f;
}
__device__ __forceinline__ bf16_t bf_zero() {
  unsigned short z = 0; bf16_t o; __builtin_memcpy(&o, &z, 2); return o;
}
__device__ __forceinline__ float sigmoidf_(float x) { return 1.f / (1.f + __expf(-x)); }

// ---- A fragment: 16 bf16 weights along contiguous padded K -----------------
// CDNA5 ISA 7.12.2 A layout: elems 0..7 -> K = ko + half*8 + e,
//                            elems 8..15 -> K = ko + 16 + half*8 + e.
__device__ __forceinline__ v16bf load_a_frag(const bf16_t* __restrict__ wr,
                                             int ko, int half)
{
  const v8bf q0 = *(const v8bf*)(wr + ko + half * 8);        // 16B aligned
  const v8bf q1 = *(const v8bf*)(wr + ko + 16 + half * 8);
  v16bf a;
#pragma unroll
  for (int e = 0; e < 8; ++e) { a[e] = q0[e]; a[e + 8] = q1[e]; }
  return a;
}

// ---- B fragment: im2col gather; lane owns one output column ----------------
// B layout: elems 0..7 -> K = ko + half*8 + e, elems 8..15 -> K = ko + 16 + half*8 + (e-8), column = lane%16.
template <int KH, int KW, int PAD>
__device__ __forceinline__ v16bf load_b_frag(const bf16_t* __restrict__ inb,
                                             int ko, int Ktot, int half,
                                             int oh, int ow, int Hin, int Win, int HWi)
{
  constexpr int KHW = KH * KW;
  const bf16_t z = bf_zero();
  v16bf b;
  const bool full = (ko + 32 <= Ktot);     // only false on c1's padded tail
#pragma unroll
  for (int e = 0; e < 16; ++e) {
    int k = ko + half * 8 + e + (e >= 8 ? 8 : 0);
    bf16_t v = z;
    if (full || k < Ktot) {
      int ci = k / KHW;                    // compile-time divisor
      int r  = k - ci * KHW;
      int kh = r / KW;                     // compile-time divisor
      int kw = r - kh * KW;
      int ih = oh + kh - PAD;
      int iw = ow + kw - PAD;
      bool ok = (KH == 1 && KW == 1 && PAD == 0)
                  ? true
                  : ((unsigned)ih < (unsigned)Hin && (unsigned)iw < (unsigned)Win);
      if (ok) v = inb[(size_t)ci * HWi + ih * Win + iw];
    }
    b[e] = v;
  }
  return b;
}

// ---------------------------------------------------------------------------
// Implicit-GEMM conv. M = Cout (MT x 16 tiles per wave), N = bt*Hout*Wout
// (16 per wave, 8 waves/block -> 128/block), K = Cin*KH*KW in steps of 32.
// Weights: bf16, row stride Kpad = ceil32(Cin*KH*KW), zero tail.
// ---------------------------------------------------------------------------
template <int KH, int KW, int PAD, int MT>
__global__ __launch_bounds__(256)
void conv_wmma(const bf16_t* __restrict__ in, const bf16_t* __restrict__ w,
               const float* __restrict__ bias,
               bf16_t* __restrict__ out_bf, float* __restrict__ out_f32,
               int Cin, int Hin, int Win, int Cout, int Hout, int Wout, int relu)
{
  constexpr int KHW = KH * KW;
  const int lane = threadIdx.x & 31;
  const int wave = threadIdx.x >> 5;
  const int half = lane >> 4;          // 0 or 1
  const int m    = lane & 15;
  const int HWo  = Hout * Wout;
  const int HWi  = Hin * Win;
  const int Ktot = Cin * KHW;
  const int Kpad = (Ktot + 31) & ~31;

  const int n   = blockIdx.x * 128 + wave * 16 + m;   // output column
  const int co0 = blockIdx.y * (MT * 16);             // MT M-tiles

  const int bt = n / HWo;
  const int hw = n - bt * HWo;
  const int oh = hw / Wout;
  const int ow = hw - oh * Wout;

  const bf16_t* __restrict__ inb = in + (size_t)bt * Cin * HWi;
  const bf16_t* __restrict__ wbase = w + (size_t)(co0 + m) * Kpad;

  v8f acc[MT];
#pragma unroll
  for (int t = 0; t < MT; ++t) acc[t] = (v8f){};

  for (int ko = 0; ko < Kpad; ko += 32) {
    __builtin_prefetch(wbase + ko + 128, 0, 1);
    v16bf b = load_b_frag<KH, KW, PAD>(inb, ko, Ktot, half, oh, ow, Hin, Win, HWi);
#pragma unroll
    for (int t = 0; t < MT; ++t) {
      v16bf a = load_a_frag(wbase + (size_t)t * 16 * Kpad, ko, half);
      acc[t] = __builtin_amdgcn_wmma_f32_16x16x32_bf16(false, a, false, b,
                                                       (short)0, acc[t], false, false);
    }
  }

  // ---- epilogue: bias (+relu) and store ----------------------------------
#pragma unroll
  for (int t = 0; t < MT; ++t) {
#pragma unroll
    for (int r = 0; r < 8; ++r) {
      int co = co0 + t * 16 + half * 8 + r;        // C/D row = half*8 + r
      float v = acc[t][r] + bias[co];
      if (relu) v = v > 0.f ? v : 0.f;
      size_t oidx = ((size_t)bt * Cout + co) * HWo + hw;
      if (out_bf)  out_bf[oidx]  = f2bf(v);
      if (out_f32) out_f32[oidx] = v;
    }
  }
}

// ---------------------------------------------------------------------------
// Weight cast: f32 [rows][Ktot] -> bf16 [rows][Kpad], zero-padded tail.
// ---------------------------------------------------------------------------
__global__ void cast_weight_kernel(const float* __restrict__ src, bf16_t* __restrict__ dst,
                                   int Ktot, int Kpad, int total)
{
  int i = blockIdx.x * blockDim.x + threadIdx.x;
  if (i >= total) return;
  int row = i / Kpad;
  int k   = i - row * Kpad;
  dst[i] = (k < Ktot) ? f2bf(src[(size_t)row * Ktot + k]) : bf_zero();
}

// ---------------------------------------------------------------------------
// Pack x1..x6 [128,4,8,9] f32  ->  xs [768,4,72] bf16  (s = b*6 + t)
// ---------------------------------------------------------------------------
__global__ void pack_x_kernel(const float* __restrict__ x1, const float* __restrict__ x2,
                              const float* __restrict__ x3, const float* __restrict__ x4,
                              const float* __restrict__ x5, const float* __restrict__ x6,
                              bf16_t* __restrict__ out)
{
  int i = blockIdx.x * blockDim.x + threadIdx.x;
  if (i >= BT * 288) return;
  int s = i / 288, r = i - s * 288;
  int b = s / TSTEP, t = s - b * TSTEP;
  const float* p;
  switch (t) { case 0: p = x1; break; case 1: p = x2; break; case 2: p = x3; break;
               case 3: p = x4; break; case 4: p = x5; break; default: p = x6; }
  out[i] = f2bf(p[b * 288 + r]);
}

// ---------------------------------------------------------------------------
// fpin[bt][0:256][hw]   = c3[bt][c][hw]
// fpin[bt][256:320][hw] = maxpool2x2(c8[bt][c-256]) broadcast over all hw
// c8 map is 2x3 row-major; 2x2/stride2 VALID window -> max of idx {0,1,3,4}.
// ---------------------------------------------------------------------------
__global__ void build_fpin_kernel(const bf16_t* __restrict__ c3,
                                  const bf16_t* __restrict__ c8,
                                  bf16_t* __restrict__ fpin)
{
  int i = blockIdx.x * blockDim.x + threadIdx.x;
  const int CHW = 320 * 42;
  if (i >= BT * CHW) return;
  int bt = i / CHW, r = i - bt * CHW;
  int c = r / 42;
  if (c < 256) {
    int hw = r - c * 42;
    fpin[i] = c3[((size_t)bt * 256 + c) * 42 + hw];
  } else {
    const bf16_t* q = c8 + ((size_t)bt * 64 + (c - 256)) * 6;
    float mx = fmaxf(fmaxf(bf2f(q[0]), bf2f(q[1])), fmaxf(bf2f(q[3]), bf2f(q[4])));
    fpin[i] = f2bf(mx);
  }
}

// ---------------------------------------------------------------------------
// Capsule squash + 3 iterations of dynamic routing, one block per sample.
// u and routing logits live in LDS; u_hat is recomputed (cheap) per iteration.
// ---------------------------------------------------------------------------
__global__ __launch_bounds__(256)
void routing_kernel(const float* __restrict__ pc, const float* __restrict__ capsW,
                    float* __restrict__ vout)
{
  __shared__ float u_sh[NPCAP][8];     // 43 KB
  __shared__ float blog[NPCAP][2];     // 10.5 KB
  __shared__ float ssum[2][16];
  __shared__ float vsh[2][16];

  const int bt  = blockIdx.x;
  const int tid = threadIdx.x;
  const float* __restrict__ p = pc + (size_t)bt * 256 * 42;

  // load + squash:  u[hw*32+g][i] = squash8( pc[bt][g*8+i][hw] )
  for (int n = tid; n < NPCAP; n += 256) {
    int hw = n >> 5, g = n & 31;
    float s[8], sq = 0.f;
#pragma unroll
    for (int i = 0; i < 8; ++i) { s[i] = p[(g * 8 + i) * 42 + hw]; sq += s[i] * s[i]; }
    float f = (sq / (1.f + sq)) * rsqrtf(sq + 1e-9f);
#pragma unroll
    for (int i = 0; i < 8; ++i) u_sh[n][i] = s[i] * f;
    blog[n][0] = 0.f; blog[n][1] = 0.f;
  }
  __syncthreads();

  for (int it = 0; it < 3; ++it) {
    if (tid < 32) ssum[tid >> 4][tid & 15] = 0.f;
    __syncthreads();

    float local[2][16];
#pragma unroll
    for (int c = 0; c < 2; ++c)
#pragma unroll
      for (int o = 0; o < 16; ++o) local[c][o] = 0.f;

    for (int n = tid; n < NPCAP; n += 256) {
      float b0 = blog[n][0], b1 = blog[n][1];
      float mx = fmaxf(b0, b1);
      float e0 = __expf(b0 - mx), e1 = __expf(b1 - mx);
      float inv = 1.f / (e0 + e1);
      float cc0 = e0 * inv, cc1 = e1 * inv;
      const float* __restrict__ W = capsW + (size_t)n * 256;  // [2][16][8]
      float uu[8];
#pragma unroll
      for (int i = 0; i < 8; ++i) uu[i] = u_sh[n][i];
#pragma unroll
      for (int c = 0; c < 2; ++c) {
        float cc = c ? cc1 : cc0;
#pragma unroll
        for (int o = 0; o < 16; ++o) {
          float uh = 0.f;
#pragma unroll
          for (int i = 0; i < 8; ++i) uh += W[(c * 16 + o) * 8 + i] * uu[i];
          local[c][o] += cc * uh;
        }
      }
    }
#pragma unroll
    for (int c = 0; c < 2; ++c)
#pragma unroll
      for (int o = 0; o < 16; ++o) atomicAdd(&ssum[c][o], local[c][o]);
    __syncthreads();

    if (tid < 2) {                      // squash s -> v per class capsule
      float sq = 0.f;
#pragma unroll
      for (int o = 0; o < 16; ++o) sq += ssum[tid][o] * ssum[tid][o];
      float f = (sq / (1.f + sq)) * rsqrtf(sq + 1e-9f);
#pragma unroll
      for (int o = 0; o < 16; ++o) vsh[tid][o] = ssum[tid][o] * f;
    }
    __syncthreads();

    if (it < 2) {                       // blog += <u_hat, v>
      for (int n = tid; n < NPCAP; n += 256) {
        const float* __restrict__ W = capsW + (size_t)n * 256;
        float uu[8];
#pragma unroll
        for (int i = 0; i < 8; ++i) uu[i] = u_sh[n][i];
#pragma unroll
        for (int c = 0; c < 2; ++c) {
          float acc = 0.f;
#pragma unroll
          for (int o = 0; o < 16; ++o) {
            float uh = 0.f;
#pragma unroll
            for (int i = 0; i < 8; ++i) uh += W[(c * 16 + o) * 8 + i] * uu[i];
            acc += uh * vsh[c][o];
          }
          blog[n][c] += acc;
        }
      }
      __syncthreads();
    }
  }
  if (tid < 32) vout[(size_t)bt * 32 + tid] = vsh[tid >> 4][tid & 15];
}

// hseq = sigmoid(gate_pre) * tanh(enc_pre)
__global__ void gate_kernel(const float* __restrict__ gpre, const float* __restrict__ epre,
                            float* __restrict__ hseq, int n)
{
  int i = blockIdx.x * blockDim.x + threadIdx.x;
  if (i < n) hseq[i] = sigmoidf_(gpre[i]) * tanhf(epre[i]);
}

__global__ void zero_kernel(float* __restrict__ p, int n)
{
  int i = blockIdx.x * blockDim.x + threadIdx.x;
  if (i < n) p[i] = 0.f;
}

// k_new = sigmoid(k_old @ wkw^T + hseq[:,t] @ uw^T + stb)
__global__ __launch_bounds__(256)
void scan_step_kernel(const float* __restrict__ kin, const float* __restrict__ hseq, int t,
                      const float* __restrict__ wkw, const float* __restrict__ uw,
                      const float* __restrict__ stb, float* __restrict__ kout)
{
  int idx = blockIdx.x * 256 + threadIdx.x;       // 128*256
  int b = idx >> 8, h = idx & 255;
  const float* __restrict__ kb = kin + (size_t)b * HID;
  const float* __restrict__ hb = hseq + ((size_t)b * TSTEP + t) * HID;
  const float* __restrict__ wr = wkw + (size_t)h * HID;
  const float* __restrict__ ur = uw + (size_t)h * HID;
  float s = stb[h];
  for (int j = 0; j < HID; ++j) s += kb[j] * wr[j] + ur[j] * hb[j];
  kout[idx] = sigmoidf_(s);
}

// class logits (tid<256) + domain head (tid 256..383)
__global__ __launch_bounds__(256)
void head_kernel(const float* __restrict__ kfin, const float* __restrict__ clw,
                 const float* __restrict__ clb, const float* __restrict__ v,
                 const float* __restrict__ d1w, const float* __restrict__ d1b,
                 const float* __restrict__ d2w, const float* __restrict__ d2b,
                 float* __restrict__ out)
{
  int tid = blockIdx.x * blockDim.x + threadIdx.x;
  if (tid < 256) {                                  // class_logits [128,2]
    int b = tid >> 1, c = tid & 1;
    float s = clb[c];
    for (int j = 0; j < HID; ++j) s += kfin[b * HID + j] * clw[c * HID + j];
    out[b * 2 + c] = s;
  } else if (tid < 384) {                           // domain_logits [128,2]
    int b = tid - 256;
    float emo[32];
#pragma unroll
    for (int j = 0; j < 32; ++j) {
      float s = 0.f;
      for (int t2 = 0; t2 < TSTEP; ++t2) s += v[((size_t)b * TSTEP + t2) * 32 + j];
      emo[j] = s * (1.f / 6.f);
    }
    float o0 = d2b[0], o1 = d2b[1];
    for (int hh = 0; hh < 128; ++hh) {
      float a = d1b[hh];
#pragma unroll
      for (int j = 0; j < 32; ++j) a += emo[j] * d1w[hh * 32 + j];
      a = a > 0.f ? a : 0.f;
      o0 += a * d2w[hh];
      o1 += a * d2w[128 + hh];
    }
    out[256 + b * 2 + 0] = o0;
    out[256 + b * 2 + 1] = o1;
  }
}

// ---------------------------------------------------------------------------
extern "C" void kernel_launch(void* const* d_in, const int* in_sizes, int n_in,
                              void* d_out, int out_size, void* d_ws, size_t ws_size,
                              hipStream_t stream)
{
  (void)in_sizes; (void)n_in; (void)out_size; (void)ws_size;
  const float* x[6];
  for (int i = 0; i < 6; ++i) x[i] = (const float*)d_in[i];
  const float *c1w = (const float*)d_in[6],  *c1b = (const float*)d_in[7];
  const float *c2w = (const float*)d_in[8],  *c2b = (const float*)d_in[9];
  const float *c3w = (const float*)d_in[10], *c3b = (const float*)d_in[11];
  const float *c4w = (const float*)d_in[12], *c4b = (const float*)d_in[13];
  const float *c5w = (const float*)d_in[14], *c5b = (const float*)d_in[15];
  const float *c6w = (const float*)d_in[16], *c6b = (const float*)d_in[17];
  const float *c7w = (const float*)d_in[18], *c7b = (const float*)d_in[19];
  const float *c8w = (const float*)d_in[20], *c8b = (const float*)d_in[21];
  const float *fpw = (const float*)d_in[22], *fpb = (const float*)d_in[23];
  const float *pcw = (const float*)d_in[24], *pcb = (const float*)d_in[25];
  const float *capsW = (const float*)d_in[26];
  const float *d1w = (const float*)d_in[27], *d1b = (const float*)d_in[28];
  const float *d2w = (const float*)d_in[29], *d2b = (const float*)d_in[30];
  const float *selw = (const float*)d_in[31], *selb = (const float*)d_in[32];
  const float *encw = (const float*)d_in[33], *encb = (const float*)d_in[34];
  const float *wkw = (const float*)d_in[35], *uw = (const float*)d_in[36];
  const float *stb = (const float*)d_in[37];
  const float *clw = (const float*)d_in[38], *clb = (const float*)d_in[39];

  // ---- workspace bump allocator (deterministic) ---------------------------
  char* ws = (char*)d_ws;
  size_t off = 0;
  auto alloc = [&](size_t bytes) -> char* {
    char* p = ws + off;
    off += (bytes + 255) & ~(size_t)255;
    return p;
  };
  // activations
  bf16_t* xbf   = (bf16_t*)alloc((size_t)BT * 4 * 72 * 2);
  bf16_t* a1    = (bf16_t*)alloc((size_t)BT * 64 * 72 * 2);
  bf16_t* a2    = (bf16_t*)alloc((size_t)BT * 128 * 56 * 2);
  bf16_t* a3    = (bf16_t*)alloc((size_t)BT * 256 * 42 * 2);
  bf16_t* a4    = (bf16_t*)alloc((size_t)BT * 512 * 30 * 2);
  bf16_t* a5    = (bf16_t*)alloc((size_t)BT * 1024 * 20 * 2);
  bf16_t* a6    = (bf16_t*)alloc((size_t)BT * 512 * 12 * 2);
  bf16_t* a7    = (bf16_t*)alloc((size_t)BT * 256 * 6 * 2);
  bf16_t* a8    = (bf16_t*)alloc((size_t)BT * 64 * 6 * 2);
  bf16_t* fpin  = (bf16_t*)alloc((size_t)BT * 320 * 42 * 2);
  bf16_t* pyr   = (bf16_t*)alloc((size_t)BT * 512 * 42 * 2);
  float*  pcout = (float*)alloc((size_t)BT * 256 * 42 * 4);
  float*  vbuf  = (float*)alloc((size_t)BT * 32 * 4);
  float*  gpre  = (float*)alloc((size_t)BT * HID * 4);
  float*  epre  = (float*)alloc((size_t)BT * HID * 4);
  float*  hs    = (float*)alloc((size_t)BT * HID * 4);
  float*  kA    = (float*)alloc((size_t)BSZ * HID * 4);
  float*  kB    = (float*)alloc((size_t)BSZ * HID * 4);

  dim3 blk(256);

  // bf16 weight buffers ([Cout][Kpad], zero tail) + cast launches
  auto castw = [&](const float* src, int Cout, int Ktot) -> bf16_t* {
    int Kpad = (Ktot + 31) & ~31;
    int total = Cout * Kpad;
    bf16_t* dst = (bf16_t*)alloc((size_t)total * 2);
    cast_weight_kernel<<<(total + 255) / 256, blk, 0, stream>>>(src, dst, Ktot, Kpad, total);
    return dst;
  };
  bf16_t* c1wb = castw(c1w,   64,    4 * 25);
  bf16_t* c2wb = castw(c2w,  128,   64 * 16);
  bf16_t* c3wb = castw(c3w,  256,  128 * 16);
  bf16_t* c4wb = castw(c4w,  512,  256 * 16);
  bf16_t* c5wb = castw(c5w, 1024,  512 * 16);
  bf16_t* c6wb = castw(c6w,  512, 1024 * 16);
  bf16_t* c7wb = castw(c7w,  256,  512 * 16);
  bf16_t* c8wb = castw(c8w,   64,  256);
  bf16_t* fpwb = castw(fpw,  512,  320);
  bf16_t* pcwb = castw(pcw,  256,  512 * 9);
  bf16_t* selwb = castw(selw, HID, 512 * 42);
  bf16_t* encwb = castw(encw, HID, 512 * 42);

#define CONV(KHh, KWw, PADp, MTt, in_, w_, b_, obf_, of32_, Cin_, Hin_, Win_, Cout_, Hout_, Wout_, relu_) \
  do {                                                                                                   \
    int Ntot = BT * (Hout_) * (Wout_);                                                                   \
    dim3 grid(Ntot / 128, (Cout_) / ((MTt) * 16));                                                       \
    conv_wmma<KHh, KWw, PADp, MTt><<<grid, blk, 0, stream>>>(in_, w_, b_, obf_, of32_,                   \
        Cin_, Hin_, Win_, Cout_, Hout_, Wout_, relu_);                                                   \
  } while (0)

  // pack inputs -> bf16
  pack_x_kernel<<<(BT * 288 + 255) / 256, blk, 0, stream>>>(
      x[0], x[1], x[2], x[3], x[4], x[5], xbf);

  // conv pyramid (MT=8 wherever Cout % 128 == 0)
  CONV(5, 5, 2, 4, xbf, c1wb, c1b, a1, (float*)nullptr,    4, 8, 9,   64, 8, 9, 1);
  CONV(4, 4, 1, 8, a1,  c2wb, c2b, a2, (float*)nullptr,   64, 8, 9,  128, 7, 8, 1);
  CONV(4, 4, 1, 8, a2,  c3wb, c3b, a3, (float*)nullptr,  128, 7, 8,  256, 6, 7, 1);
  CONV(4, 4, 1, 8, a3,  c4wb, c4b, a4, (float*)nullptr,  256, 6, 7,  512, 5, 6, 1);
  CONV(4, 4, 1, 8, a4,  c5wb, c5b, a5, (float*)nullptr,  512, 5, 6, 1024, 4, 5, 1);
  CONV(4, 4, 1, 8, a5,  c6wb, c6b, a6, (float*)nullptr, 1024, 4, 5,  512, 3, 4, 1);
  CONV(4, 4, 1, 8, a6,  c7wb, c7b, a7, (float*)nullptr,  512, 3, 4,  256, 2, 3, 1);
  CONV(1, 1, 0, 4, a7,  c8wb, c8b, a8, (float*)nullptr,  256, 2, 3,   64, 2, 3, 1);

  // maxpool(c8) broadcast + concat with c3 -> fpin [BT,320,42]
  build_fpin_kernel<<<(BT * 320 * 42 + 255) / 256, blk, 0, stream>>>(a3, a8, fpin);

  // feature pyramid 1x1 conv -> pyr [BT,512,42]  (flat features = pyr.reshape)
  CONV(1, 1, 0, 8, fpin, fpwb, fpb, pyr, (float*)nullptr, 320, 6, 7, 512, 6, 7, 1);

  // primary capsule conv (no relu, f32 out for squash precision)
  CONV(3, 3, 1, 8, pyr, pcwb, pcb, (bf16_t*)nullptr, pcout, 512, 6, 7, 256, 6, 7, 0);

  // capsule routing -> v [BT,2,16]
  routing_kernel<<<BT, blk, 0, stream>>>(pcout, capsW, vbuf);

  // sel/enc big GEMMs as 1x1 "convs": [768,21504] x [21504,256]
  CONV(1, 1, 0, 8, pyr, selwb, selb, (bf16_t*)nullptr, gpre, 512 * 42, 1, 1, HID, 1, 1, 0);
  CONV(1, 1, 0, 8, pyr, encwb, encb, (bf16_t*)nullptr, epre, 512 * 42, 1, 1, HID, 1, 1, 0);

  // gated hidden sequence
  gate_kernel<<<(BT * HID + 255) / 256, blk, 0, stream>>>(gpre, epre, hs, BT * HID);

  // recurrent scan over T
  zero_kernel<<<(BSZ * HID + 255) / 256, blk, 0, stream>>>(kA, BSZ * HID);
  float* kcur = kA;
  float* knxt = kB;
  for (int t = 0; t < TSTEP; ++t) {
    scan_step_kernel<<<BSZ, blk, 0, stream>>>(kcur, hs, t, wkw, uw, stb, knxt);
    float* tmp = kcur; kcur = knxt; knxt = tmp;
  }

  // heads -> d_out: class_logits[128,2] ++ domain_logits[128,2]
  head_kernel<<<2, blk, 0, stream>>>(kcur, clw, clb, vbuf, d1w, d1b, d2w, d2b,
                                     (float*)d_out);
}
